// FusedGlobalMutualInformationLoss_86346022519518
// MI455X (gfx1250) — matrix operand order repaired and checked by reading, not compiled, not run on hardware
//
#include <hip/hip_runtime.h>
#include <hip/hip_bf16.h>

// CDNA5 / gfx1250 wave32 WMMA types
typedef __attribute__((ext_vector_type(16))) _Float16 v16h;
typedef __attribute__((ext_vector_type(8)))  _Float16 v8h;
typedef __attribute__((ext_vector_type(2)))  _Float16 h2;
typedef __attribute__((ext_vector_type(8)))  float    v8f;

#define NBINS 32
// preterm*(x-c_k)^2 == 0.5*(31x-k)^2 for range [0,1], 32 bins, sigma_ratio=1
#define EXP2_SCALE 0.7213475204444817f   // 0.5 * log2(e)
#define SMOOTH 1e-7f

__device__ __forceinline__ v16h frag_cat(v8h a, v8h b) {
  return __builtin_shufflevector(a, b, 0,1,2,3,4,5,6,7,8,9,10,11,12,13,14,15);
}

// Normalized Parzen weights for two voxels -> packed h2 transposed store
// (dst already offset by this lane's element pair; row stride 64 halves).
__device__ __forceinline__ void parzen2_store(float x0, float x1, _Float16* dst) {
  float u0 = 31.0f * x0, u1 = 31.0f * x1;
  float w0[32], w1[32], s0 = 0.f, s1 = 0.f;
#pragma unroll
  for (int k = 0; k < 32; ++k) {
    float d0 = u0 - (float)k, d1 = u1 - (float)k;
    float e0 = exp2f(-EXP2_SCALE * d0 * d0);   // v_exp_f32
    float e1 = exp2f(-EXP2_SCALE * d1 * d1);
    w0[k] = e0; s0 += e0;
    w1[k] = e1; s1 += e1;
  }
  float i0 = 1.0f / s0, i1 = 1.0f / s1;
#pragma unroll
  for (int k = 0; k < 32; ++k) {
    h2 p; p.x = (_Float16)(w0[k] * i0); p.y = (_Float16)(w1[k] * i1);
    *(h2*)(dst + k * 64) = p;                  // one ds_store_b32 per bin
  }
}

// Kernel 1: per-block partial 32x32 joint histogram via v_wmma_f32_16x16x32_f16.
// pab[k,l] = sum_n wa[n,k]*wb[n,l];  A = wa^T (16x32 f16 tiles), B = wb (32x16 f16 tiles).
// Each wave processes 64 voxels per chunk (2 per lane), K=64 = two K=32 WMMA groups.
__global__ __launch_bounds__(256) void mi_joint_hist_kernel(
    const float* __restrict__ pred, const float* __restrict__ targ,
    float* __restrict__ partial, int N)
{
  // 8 waves * 8KB private staging: [2 images][32 bins][64 elems] f16, transposed.
  __shared__ __align__(16) unsigned char smem[8 * 8192];
  const int lane = threadIdx.x & 31;
  const int wave = threadIdx.x >> 5;
  _Float16* waT = (_Float16*)(smem + wave * 8192);   // [32][64] halves (pred)
  _Float16* wbT = waT + 2048;                        // [32][64] halves (target)

  const int col  = lane & 15;   // bin within 16-tile (M for A, N for B)
  const int hi16 = lane >> 4;   // lane group selects K-subrange per ISA layout
  // f16 A 16x32: lanes 0-15 hold K={0..7,16..23}, lanes 16-31 K={8..15,24..31}
  const int aoff = hi16 * 8;    // halves
  // f16 B 32x16: lanes 0-15 hold K={0..7,16..23}, lanes 16-31 K={8..15,24..31}
  const int boff = hi16 * 8;    // halves
  const _Float16* rowA0 = waT + (col     ) * 64;
  const _Float16* rowA1 = waT + (16 + col) * 64;
  const _Float16* rowB0 = wbT + (col     ) * 64;
  const _Float16* rowB1 = wbT + (16 + col) * 64;

  v8f c00 = {}, c01 = {}, c10 = {}, c11 = {};

  const int nchunk  = N >> 6;                 // 64 voxels per wave-chunk
  const int gwave   = blockIdx.x * 8 + wave;
  const int wstride = gridDim.x * 8;
  const float2* __restrict__ pred2 = (const float2*)pred;
  const float2* __restrict__ targ2 = (const float2*)targ;

  for (int ch = gwave; ch < nchunk; ch += wstride) {
    const int pidx = (ch << 5) + lane;        // float2 index: voxels 2*pidx, 2*pidx+1
    const float2 x = pred2[pidx];
    const float2 y = targ2[pidx];
    __builtin_prefetch(pred2 + pidx + (wstride << 5), 0, 1);  // global_prefetch_b8
    __builtin_prefetch(targ2 + pidx + (wstride << 5), 0, 1);

    parzen2_store(x.x, x.y, waT + 2 * lane);
    parzen2_store(y.x, y.y, wbT + 2 * lane);

    // wave-private staging: only need intra-wave store->load ordering
    asm volatile("s_wait_dscnt 0" ::: "memory");

#pragma unroll
    for (int g = 0; g < 2; ++g) {             // two K=32 groups (elements g*32..+31)
      const int gb = g * 32;                  // halves
      v16h a0 = frag_cat(*(const v8h*)(rowA0 + gb + aoff), *(const v8h*)(rowA0 + gb + 16 + aoff));
      v16h a1 = frag_cat(*(const v8h*)(rowA1 + gb + aoff), *(const v8h*)(rowA1 + gb + 16 + aoff));
      v16h b0 = frag_cat(*(const v8h*)(rowB0 + gb + boff), *(const v8h*)(rowB0 + gb + 16 + boff));
      v16h b1 = frag_cat(*(const v8h*)(rowB1 + gb + boff), *(const v8h*)(rowB1 + gb + 16 + boff));

      c00 = __builtin_amdgcn_wmma_f32_16x16x32_f16(false, a0, false, b0, (short)0, c00, false, false);
      c01 = __builtin_amdgcn_wmma_f32_16x16x32_f16(false, a0, false, b1, (short)0, c01, false, false);
      c10 = __builtin_amdgcn_wmma_f32_16x16x32_f16(false, a1, false, b0, (short)0, c10, false, false);
      c11 = __builtin_amdgcn_wmma_f32_16x16x32_f16(false, a1, false, b1, (short)0, c11, false, false);
    }

    asm volatile("" ::: "memory");  // keep next iter's stores after this iter's loads
  }

  // Spill this wave's 32x32 f32 tile into its own (reused) LDS region.
  asm volatile("" ::: "memory");
  float* tile = (float*)(smem + wave * 8192);  // first 4KB of the wave region
#pragma unroll
  for (int r = 0; r < 8; ++r) {
    // C/D layout: VGPR r, lanes 0-15 -> M=r, lanes 16-31 -> M=r+8; N = lane&15
    int row = r + hi16 * 8;
    tile[(row     ) * 32 +      col] = c00[r];
    tile[(row     ) * 32 + 16 + col] = c01[r];
    tile[(16 + row) * 32 +      col] = c10[r];
    tile[(16 + row) * 32 + 16 + col] = c11[r];
  }
  __syncthreads();

  // Deterministic fixed-order reduction of the 8 wave tiles -> block partial.
  float* outp = partial + (size_t)blockIdx.x * 1024;
#pragma unroll
  for (int j = 0; j < 4; ++j) {
    int cell = threadIdx.x * 4 + j;
    float s = 0.f;
#pragma unroll
    for (int w = 0; w < 8; ++w) s += ((const float*)smem)[w * 2048 + cell];
    outp[cell] = s;
  }
}

// Kernel 2: fold block partials (fixed order), marginals from the joint
// (pa[k]=sum_l pab[k,l] is exact since weights are normalized), MI log term,
// deterministic tree reduce, write -mi.
__global__ __launch_bounds__(1024) void mi_finalize_kernel(
    const float* __restrict__ partial, float* __restrict__ out, int N, int nblocks)
{
  __shared__ float spab[1024];
  __shared__ float spa[32], spb[32];
  __shared__ float red[1024];
  const int t = threadIdx.x;   // cell id: k = t>>5, l = t&31

  float s = 0.f;
  for (int b = 0; b < nblocks; ++b) s += partial[(size_t)b * 1024 + t];
  float pab = s / (float)N;
  spab[t] = pab;
  __syncthreads();

  if (t < 32) {
    float a = 0.f;
    for (int l = 0; l < 32; ++l) a += spab[t * 32 + l];
    spa[t] = a;
  } else if (t < 64) {
    int c = t - 32; float a = 0.f;
    for (int k = 0; k < 32; ++k) a += spab[k * 32 + c];
    spb[c] = a;
  }
  __syncthreads();

  const int k = t >> 5, l = t & 31;
  float papb = spa[k] * spb[l];
  red[t] = spab[t] * logf((spab[t] + SMOOTH) / (papb + SMOOTH) + SMOOTH);
  __syncthreads();

  for (int off = 512; off > 0; off >>= 1) {
    if (t < off) red[t] += red[t + off];
    __syncthreads();
  }
  if (t == 0) out[0] = -red[0];
}

extern "C" void kernel_launch(void* const* d_in, const int* in_sizes, int n_in,
                              void* d_out, int out_size, void* d_ws, size_t ws_size,
                              hipStream_t stream) {
  const float* pred = (const float*)d_in[0];
  const float* targ = (const float*)d_in[1];
  float* ws  = (float*)d_ws;
  float* out = (float*)d_out;
  const int N = in_sizes[0];           // 128^3 = 2,097,152 (divisible by 64)

  // one 32x32 f32 partial (4KB) per block; clamp by available scratch
  size_t cap = ws_size / (1024 * sizeof(float));
  int nblocks = (int)(cap < 1 ? 1 : (cap > 256 ? 256 : cap));
  int nchunk = N >> 6;
  if (nblocks * 8 > nchunk && nchunk >= 8) nblocks = nchunk / 8;
  if (nblocks < 1) nblocks = 1;

  mi_joint_hist_kernel<<<nblocks, 256, 0, stream>>>(pred, targ, ws, N);
  mi_finalize_kernel<<<1, 1024, 0, stream>>>(ws, out, N, nblocks);
}
